// SRED_rep_rho_DO_78159814852738
// MI455X (gfx1250) — matrix operand order repaired and checked by reading, not compiled, not run on hardware
//
#include <hip/hip_runtime.h>
#include <hip/hip_bf16.h>
#include <math.h>

// Problem constants (from reference)
#define BB   32
#define LS   128
#define MM   16
#define LW   128
#define LY   128
#define NSTEP 10
#define HID  256
#define DIN  640   // 2*LS + 2*LW + LY

// Workspace layout (float offsets). Total 473600 floats = ~1.85 MB.
#define OFF_PHI    0         // B*LS        = 4096
#define OFF_SRE    4096      // B*LS
#define OFF_SIM    8192      // B*LS
#define OFF_GSRE   12288     // B*LS*M      = 65536
#define OFF_GSIM   77824
#define OFF_HSRE   143360
#define OFF_HSIM   208896
#define OFF_SGSRE  274432    // B*M = 512
#define OFF_SGSIM  274944
#define OFF_SHSRE  275456
#define OFF_SHSIM  275968
#define OFF_ETA    276480    // B*LS*M
#define OFF_H      342016    // 512*256 = 131072
#define OFF_RHO    473088    // 512

typedef __attribute__((ext_vector_type(16))) __bf16 v16bf;
typedef __attribute__((ext_vector_type(8)))  float  v8f;
typedef __attribute__((ext_vector_type(4)))  unsigned int u32x4;
typedef __attribute__((ext_vector_type(8)))  int   i32x8;
typedef __attribute__((ext_vector_type(4)))  int   i32x4;

// ---------------------------------------------------------------- TDM helper
// Issue a 2D tensor DMA: `rows` rows of `row_elems` f32, row stride
// `row_stride_elems` f32, from global `gaddr` into LDS byte address `lds_addr`.
// D# layout per CDNA5 ISA ch.8 (group0: count/lds/global/type, group1:
// data_size=4B, tensor dims, tile dims, strides). Wave-level op, EXEC ignored,
// tracked with TENSORcnt. This toolchain exposes the 6-arg builtin
// (g0, g1, g2, g3, extra group, cpol).
__device__ __forceinline__ void tdm_load_2d(unsigned lds_addr, const void* gaddr,
                                            unsigned row_elems, unsigned rows,
                                            unsigned row_stride_elems) {
    unsigned long long ga = (unsigned long long)gaddr;
    u32x4 g0;
    g0.x = 1u;                                   // count=1 (valid user descriptor)
    g0.y = lds_addr;                             // lds_addr [63:32]
    g0.z = (unsigned)(ga & 0xffffffffu);         // global_addr[31:0]
    g0.w = (unsigned)((ga >> 32) & 0x01ffffffu)  // global_addr[56:32]
           | (2u << 30);                         // type=2 ("image")
    i32x8 g1;
    g1[0] = (int)(2u << 16);                     // workgroup_mask=0, data_size=2 (4B)
    g1[1] = (int)((row_elems & 0xffffu) << 16);  // tensor_dim0[15:0]
    g1[2] = (int)(((row_elems >> 16) & 0xffffu)  // tensor_dim0[31:16]
           | ((rows & 0xffffu) << 16));          // tensor_dim1[15:0]
    g1[3] = (int)(((rows >> 16) & 0xffffu)
           | ((row_elems & 0xffffu) << 16));     // tile_dim0 = row_elems
    g1[4] = (int)(rows & 0xffffu);               // tile_dim1 = rows, tile_dim2=0
    g1[5] = (int)row_stride_elems;               // tensor_dim0_stride[31:0]
    g1[6] = 0;
    g1[7] = 0;
    i32x4 z4 = {0, 0, 0, 0};
    i32x8 z8 = {0, 0, 0, 0, 0, 0, 0, 0};
    __builtin_amdgcn_tensor_load_to_lds(g0, g1, z4, z4, z8, 0);
}

// ---------------------------------------------------------------- phi copy
__global__ void copy_phi_kernel(const float* __restrict__ phi_in,
                                float* __restrict__ phi) {
    int i = blockIdx.x * blockDim.x + threadIdx.x;
    if (i < BB * LS) phi[i] = phi_in[i];
}

// ---------------------------------------------------------------- s = exp(i*phi)
__global__ void s_kernel(const float* __restrict__ phi,
                         float* __restrict__ s_re, float* __restrict__ s_im,
                         float* __restrict__ out_s, int step) {
    int b = blockIdx.x, l = threadIdx.x;      // grid=B, block=LS
    float p = phi[b * LS + l];
    float sn, cs;
    __sincosf(p, &sn, &cs);
    s_re[b * LS + l] = cs;
    s_im[b * LS + l] = sn;
    float* o = out_s + ((size_t)b * (NSTEP + 1) + step) * (2 * LS) + 2 * l;
    o[0] = cs;
    o[1] = sn;
}

// ---------------------------------------------------------------- Gs/Hs stream (TDM)
// Gs[b,l,m] = sum_k G[b,l,k,m] * s[b,k]  (complex), same for H.
// grid=(B, LS/8), block=128 (16 m x 8 l). K split into 8 chunks of 16; wave 0
// drives 4 TDM DMAs per chunk (G_re/G_im/H_re/H_im 8x256-f32 tiles, stride
// 2048) into double-buffered LDS while all waves compute on the other buffer.
#define KCH   16                 // k per chunk
#define NCH   (LS / KCH)         // 8 chunks
#define TILEF (8 * KCH * MM)     // 2048 floats per array per buffer

__global__ void gshs_tdm_kernel(const float* __restrict__ G_re, const float* __restrict__ G_im,
                                const float* __restrict__ H_re, const float* __restrict__ H_im,
                                const float* __restrict__ s_re, const float* __restrict__ s_im,
                                float* __restrict__ Gs_re, float* __restrict__ Gs_im,
                                float* __restrict__ Hs_re, float* __restrict__ Hs_im) {
    __shared__ float sre[LS], sim[LS];
    __shared__ float tile[2][4][TILEF];   // 64KB double buffer
    int b = blockIdx.x;
    int tid = threadIdx.x;
    int l0 = blockIdx.y * 8;
    sre[tid] = s_re[b * LS + tid];
    sim[tid] = s_im[b * LS + tid];

    const float* srcs[4] = {G_re, G_im, H_re, H_im};

    // preload chunk 0
    if (tid < 32) {
        size_t e0 = ((size_t)(b * LS + l0) * LS) * MM;   // chunk 0, element offset
        for (int a = 0; a < 4; ++a)
            tdm_load_2d((unsigned)(size_t)(void*)&tile[0][a][0], srcs[a] + e0,
                        KCH * MM, 8, LS * MM);
        __builtin_amdgcn_s_wait_tensorcnt(0);
    }
    __syncthreads();

    int m = tid & 15, lsub = tid >> 4;
    float gsr = 0.f, gsi = 0.f, hsr = 0.f, hsi = 0.f;

    for (int c = 0; c < NCH; ++c) {
        int cur = c & 1;
        // kick off next chunk's DMAs before computing on the current buffer
        if (tid < 32 && c + 1 < NCH) {
            size_t e1 = ((size_t)(b * LS + l0) * LS + (size_t)(c + 1) * KCH) * MM;
            for (int a = 0; a < 4; ++a)
                tdm_load_2d((unsigned)(size_t)(void*)&tile[cur ^ 1][a][0], srcs[a] + e1,
                            KCH * MM, 8, LS * MM);
        }
        int base = lsub * (KCH * MM);
#pragma unroll 4
        for (int kk = 0; kk < KCH; ++kk) {
            int ke = (kk + lsub) & (KCH - 1);   // per-l rotation: bank-conflict free
            int off = base + ke * MM + m;
            float srk = sre[c * KCH + ke], sik = sim[c * KCH + ke];
            float a  = tile[cur][0][off], ci = tile[cur][1][off];
            gsr = fmaf(a, srk, fmaf(-ci, sik, gsr));
            gsi = fmaf(a, sik, fmaf( ci, srk, gsi));
            float e  = tile[cur][2][off], f  = tile[cur][3][off];
            hsr = fmaf(e, srk, fmaf(-f, sik, hsr));
            hsi = fmaf(e, sik, fmaf( f, srk, hsi));
        }
        if (tid < 32 && c + 1 < NCH) __builtin_amdgcn_s_wait_tensorcnt(0);
        __syncthreads();
    }

    size_t o = (size_t)(b * LS + l0 + lsub) * MM + m;
    Gs_re[o] = gsr; Gs_im[o] = gsi;
    Hs_re[o] = hsr; Hs_im[o] = hsi;
}

// ---------------------------------------------------------------- sGs/sHs reduce
__global__ void sgshs_kernel(const float* __restrict__ Gs_re, const float* __restrict__ Gs_im,
                             const float* __restrict__ Hs_re, const float* __restrict__ Hs_im,
                             const float* __restrict__ s_re, const float* __restrict__ s_im,
                             float* __restrict__ sGs_re, float* __restrict__ sGs_im,
                             float* __restrict__ sHs_re, float* __restrict__ sHs_im) {
    int b = blockIdx.x, tid = threadIdx.x;
    int m = tid & 15, lg = tid >> 4;
    float ar = 0.f, ai = 0.f, br = 0.f, bi = 0.f;
    for (int l = lg; l < LS; l += 8) {
        float sr = s_re[b * LS + l], si = s_im[b * LS + l];   // conj = sr - i*si
        size_t o = (size_t)(b * LS + l) * MM + m;
        __builtin_prefetch(&Gs_re[o + 8 * MM], 0, 3);
        float gr = Gs_re[o], gi = Gs_im[o];
        ar += sr * gr + si * gi;
        ai += sr * gi - si * gr;
        float hr = Hs_re[o], hi = Hs_im[o];
        br += sr * hr + si * hi;
        bi += sr * hi - si * hr;
    }
    __shared__ float red[4][8][16];
    red[0][lg][m] = ar; red[1][lg][m] = ai;
    red[2][lg][m] = br; red[3][lg][m] = bi;
    __syncthreads();
    if (lg == 0) {
        float t0 = 0.f, t1 = 0.f, t2 = 0.f, t3 = 0.f;
        for (int j = 0; j < 8; ++j) {
            t0 += red[0][j][m]; t1 += red[1][j][m];
            t2 += red[2][j][m]; t3 += red[3][j][m];
        }
        sGs_re[b * MM + m] = t0; sGs_im[b * MM + m] = t1;
        sHs_re[b * MM + m] = t2; sHs_im[b * MM + m] = t3;
    }
}

// ---------------------------------------------------------------- eta
// eta[b,l,m] = Re(2/sHs^2) * Im((sHs*Gs - sGs*Hs) * conj(s_l))
__global__ void eta_kernel(const float* __restrict__ Gs_re, const float* __restrict__ Gs_im,
                           const float* __restrict__ Hs_re, const float* __restrict__ Hs_im,
                           const float* __restrict__ sGs_re, const float* __restrict__ sGs_im,
                           const float* __restrict__ sHs_re, const float* __restrict__ sHs_im,
                           const float* __restrict__ s_re, const float* __restrict__ s_im,
                           float* __restrict__ eta) {
    int b = blockIdx.x;
    int tid = threadIdx.x;
    int m = tid & 15, lsub = tid >> 4;
    int l = blockIdx.y * 8 + lsub;
    size_t o = (size_t)(b * LS + l) * MM + m;

    float shr = sHs_re[b * MM + m], shi = sHs_im[b * MM + m];
    float sgr = sGs_re[b * MM + m], sgi = sGs_im[b * MM + m];
    float gr = Gs_re[o], gi = Gs_im[o];
    float hr = Hs_re[o], hi = Hs_im[o];
    // t = sHs*Gs - sGs*Hs
    float tr = shr * gr - shi * gi - (sgr * hr - sgi * hi);
    float ti = shr * gi + shi * gr - (sgr * hi + sgi * hr);
    // Im(t * conj(s_l))
    float sr = s_re[b * LS + l], si = s_im[b * LS + l];
    float imnum = ti * sr - tr * si;
    // Re(2/sHs^2) = 2*(a^2-b^2)/|sHs|^4
    float a2 = shr * shr, b2v = shi * shi;
    float n2 = a2 + b2v;
    float rinv = (a2 - b2v) / (n2 * n2);
    eta[o] = 2.f * rinv * imnum;
}

// ---------------------------------------------------------------- MLP GEMM (WMMA)
// h[512,256] = relu(x[512,640] @ W1[640,256] + b1), x synthesized on the fly.
__device__ __forceinline__ float x_elem(int b, int m, int c,
                                        const float* __restrict__ s_re,
                                        const float* __restrict__ s_im,
                                        const float* __restrict__ w_re,
                                        const float* __restrict__ w_im,
                                        const float* __restrict__ y_M) {
    if (c < 128)  return s_re[b * LS + c];
    if (c < 256)  return s_im[b * LS + (c - 128)];
    if (c < 384)  return w_re[(b * LW + (c - 256)) * MM + m];
    if (c < 512)  return w_im[(b * LW + (c - 384)) * MM + m];
    return y_M[(c - 512) * MM + m];
}

__global__ void mlp_wmma_kernel(const float* __restrict__ s_re, const float* __restrict__ s_im,
                                const float* __restrict__ w_re, const float* __restrict__ w_im,
                                const float* __restrict__ y_M,
                                const float* __restrict__ W1, const float* __restrict__ b1,
                                float* __restrict__ h) {
    // 512x256 out -> 512 tiles of 16x16; 1 tile/wave, 8 waves/block, 64 blocks.
    int wave = threadIdx.x >> 5;
    int lane = threadIdx.x & 31;
    int tile = blockIdx.x * 8 + wave;
    int tileRow = tile >> 4;   // 0..31
    int tileCol = tile & 15;   // 0..15
    int rlane = lane & 15;
    int grp = lane >> 4;       // lane half

    int globRow = tileRow * 16 + rlane;   // A row for this lane
    int b = globRow >> 4;
    int m = globRow & 15;
    int ncol = tileCol * 16 + rlane;      // B/C column for this lane

    v8f acc = {};
    for (int kb = 0; kb < DIN / 32; ++kb) {   // 20 k-blocks of 32
        v16bf a, bm;
        // A 16x32 bf16, k order this kernel uses: elem e -> K = e + 16*grp
        // B 32x16 bf16, same k order as A: elem e -> K = e + 16*grp, N = rlane
#pragma unroll
        for (int e = 0; e < 16; ++e) {
            int ka = e + 16 * grp;
            a[e] = (__bf16)x_elem(b, m, kb * 32 + ka, s_re, s_im, w_re, w_im, y_M);
            int kbr = e + 16 * grp;
            bm[e] = (__bf16)W1[(kb * 32 + kbr) * HID + ncol];
        }
        acc = __builtin_amdgcn_wmma_f32_16x16x32_bf16(
            false, a, false, bm, (short)0, acc, false, false);
    }
    // C/D layout: VGPR i -> row i + 8*grp, col = rlane. Fuse bias + relu.
    float bias = b1[ncol];
#pragma unroll
    for (int i = 0; i < 8; ++i) {
        int r = tileRow * 16 + i + 8 * grp;
        float v = acc[i] + bias;
        h[(size_t)r * HID + ncol] = v > 0.f ? v : 0.f;
    }
}

// ---------------------------------------------------------------- rho = h @ W2 + b2
__global__ void rho_kernel(const float* __restrict__ h,
                           const float* __restrict__ W2, const float* __restrict__ b2,
                           float* __restrict__ rho, float* __restrict__ out_rho, int step) {
    int row = blockIdx.x;
    int tid = threadIdx.x;
    __shared__ float red[HID];
    red[tid] = h[(size_t)row * HID + tid] * W2[tid];
    __syncthreads();
    for (int s = HID / 2; s > 0; s >>= 1) {
        if (tid < s) red[tid] += red[tid + s];
        __syncthreads();
    }
    if (tid == 0) {
        float r = red[0] + b2[0];
        rho[row] = r;
        int b = row >> 4, m = row & 15;
        out_rho[(b * NSTEP + step) * MM + m] = r;
    }
}

// ---------------------------------------------------------------- phi update
__global__ void phi_update_kernel(float* __restrict__ phi,
                                  const float* __restrict__ rho,
                                  const float* __restrict__ eta) {
    int b = blockIdx.x, l = threadIdx.x;
    float acc = 0.f;
#pragma unroll
    for (int m = 0; m < MM; ++m)
        acc = fmaf(rho[b * MM + m], eta[(size_t)(b * LS + l) * MM + m], acc);
    phi[b * LS + l] -= acc;
}

// ================================================================ launch
extern "C" void kernel_launch(void* const* d_in, const int* in_sizes, int n_in,
                              void* d_out, int out_size, void* d_ws, size_t ws_size,
                              hipStream_t stream) {
    const float* phi_in = (const float*)d_in[0];
    const float* w_re   = (const float*)d_in[1];
    const float* w_im   = (const float*)d_in[2];
    const float* y_M    = (const float*)d_in[3];
    const float* G_re   = (const float*)d_in[4];
    const float* G_im   = (const float*)d_in[5];
    const float* H_re   = (const float*)d_in[6];
    const float* H_im   = (const float*)d_in[7];
    const float* W1     = (const float*)d_in[8];
    const float* b1     = (const float*)d_in[9];
    const float* W2     = (const float*)d_in[10];
    const float* b2     = (const float*)d_in[11];

    float* out   = (float*)d_out;
    float* out_s = out;                                  // [B,11,LS] complex64
    float* out_rho = out + BB * (NSTEP + 1) * LS * 2;    // [B,10,M] f32

    float* ws = (float*)d_ws;   // needs ~1.85 MB
    float* phi    = ws + OFF_PHI;
    float* s_re   = ws + OFF_SRE;
    float* s_im   = ws + OFF_SIM;
    float* Gs_re  = ws + OFF_GSRE;
    float* Gs_im  = ws + OFF_GSIM;
    float* Hs_re  = ws + OFF_HSRE;
    float* Hs_im  = ws + OFF_HSIM;
    float* sGs_re = ws + OFF_SGSRE;
    float* sGs_im = ws + OFF_SGSIM;
    float* sHs_re = ws + OFF_SHSRE;
    float* sHs_im = ws + OFF_SHSIM;
    float* eta    = ws + OFF_ETA;
    float* hbuf   = ws + OFF_H;
    float* rho    = ws + OFF_RHO;

    copy_phi_kernel<<<(BB * LS + 255) / 256, 256, 0, stream>>>(phi_in, phi);

    for (int t = 0; t < NSTEP; ++t) {
        s_kernel<<<BB, LS, 0, stream>>>(phi, s_re, s_im, out_s, t);
        gshs_tdm_kernel<<<dim3(BB, LS / 8), 128, 0, stream>>>(
            G_re, G_im, H_re, H_im, s_re, s_im, Gs_re, Gs_im, Hs_re, Hs_im);
        sgshs_kernel<<<BB, 128, 0, stream>>>(
            Gs_re, Gs_im, Hs_re, Hs_im, s_re, s_im, sGs_re, sGs_im, sHs_re, sHs_im);
        eta_kernel<<<dim3(BB, LS / 8), 128, 0, stream>>>(
            Gs_re, Gs_im, Hs_re, Hs_im, sGs_re, sGs_im, sHs_re, sHs_im, s_re, s_im, eta);
        mlp_wmma_kernel<<<64, 256, 0, stream>>>(
            s_re, s_im, w_re, w_im, y_M, W1, b1, hbuf);
        rho_kernel<<<BB * MM, HID, 0, stream>>>(hbuf, W2, b2, rho, out_rho, t);
        phi_update_kernel<<<BB, LS, 0, stream>>>(phi, rho, eta);
    }
    // final s -> slot NSTEP of s_stack
    s_kernel<<<BB, LS, 0, stream>>>(phi, s_re, s_im, out_s, NSTEP);
}
